// MultiHeadAttention_91311004713222
// MI455X (gfx1250) — matrix operand order repaired and checked by reading, not compiled, not run on hardware
//
#include <hip/hip_runtime.h>

#define SEQ    4096
#define DMODEL 1024
#define NHEADS 16
#define DK     64

typedef __bf16 bf16;
typedef __attribute__((ext_vector_type(16))) __bf16 bf16x16;
typedef __attribute__((ext_vector_type(8)))  float  f32x8;
typedef unsigned int u32;
typedef __attribute__((ext_vector_type(4))) u32 u32x4;
typedef __attribute__((ext_vector_type(4))) int i32x4;
typedef __attribute__((ext_vector_type(8))) int i32x8;

#if defined(__has_builtin)
#if __has_builtin(__builtin_amdgcn_tensor_load_to_lds) && __has_builtin(__builtin_amdgcn_s_wait_tensorcnt)
#define HAVE_TDM 1
#endif
#endif
#ifndef HAVE_TDM
#define HAVE_TDM 0
#endif

__device__ inline f32x8 wmma_bf16(bf16x16 a, bf16x16 b, f32x8 c) {
  // D = A(16x32) * B(32x16) + C, f32 accumulate
  return __builtin_amdgcn_wmma_f32_16x16x32_bf16(
      /*neg_a=*/false, a, /*neg_b=*/false, b,
      /*c_mod=*/(short)0, c, /*reuse_a=*/false, /*reuse_b=*/false);
}

// A-frag (16x32, MxK) from LDS, rows [row0,row0+16), K cols [kbase,kbase+32),
// row stride ldk elements. ISA layout: lanes<16: K 0..7 then 16..23;
// lanes>=16: K 8..15 then 24..31 (per-lane row M = lane&15).
__device__ inline bf16x16 load_a_frag(const bf16* p, int row0, int kbase, int ldk) {
  int lane = threadIdx.x & 31;
  int half = lane >> 4, r = lane & 15;
  const bf16* rp = p + (size_t)(row0 + r) * ldk + kbase + half * 8;
  bf16x16 out;
#pragma unroll
  for (int i = 0; i < 8; ++i) out[i] = rp[i];
#pragma unroll
  for (int i = 0; i < 8; ++i) out[8 + i] = rp[16 + i];
  return out;
}

// B-frag (32x16, KxN). LDS holds B transposed as [N][K] with row stride ldk.
// lanes<16: N=lane, K=kbase+0..7 then 16..23 ; lanes>=16: same N, K=kbase+8..15 then 24..31.
__device__ inline bf16x16 load_b_frag(const bf16* p, int col0, int kbase, int ldk) {
  int lane = threadIdx.x & 31;
  int half = lane >> 4, n = lane & 15;
  const bf16* cp = p + (size_t)(col0 + n) * ldk + kbase + half * 8;
  bf16x16 out;
#pragma unroll
  for (int i = 0; i < 8; ++i) out[i] = cp[i];
#pragma unroll
  for (int i = 0; i < 8; ++i) out[8 + i] = cp[16 + i];
  return out;
}

#if HAVE_TDM
// Tensor Data Mover: 2D bf16 tile (tile0 x tile1 elements, row stride stride0
// elements in global) from global to contiguous LDS. One wave issues; tracked
// with TENSORcnt.
__device__ inline void tdm_load_tile_2d(u32 lds_addr, const void* gptr,
                                        u32 tdim0, u32 tdim1,
                                        u32 tile0, u32 tile1, u32 stride0) {
  unsigned long long ga = (unsigned long long)gptr;
  u32x4 g0;
  g0.x = 1u;                                   // count=1 valid descriptor
  g0.y = lds_addr;                             // LDS byte address
  g0.z = (u32)ga;                              // global_addr[31:0]
  g0.w = (u32)((ga >> 32) & 0x01FFFFFFull) | (2u << 30);  // addr[56:32] | type=2
  i32x8 g1;
  g1[0] = (int)(1u << 16);                     // data_size=1 (2 bytes)
  g1[1] = (int)(tdim0 << 16);                  // tensor_dim0[15:0]  @ bits 63:48
  g1[2] = (int)((tdim0 >> 16) | (tdim1 << 16));// dim0 hi | dim1 lo
  g1[3] = (int)((tdim1 >> 16) | (tile0 << 16));// dim1 hi | tile_dim0
  g1[4] = (int)(tile1 & 0xFFFFu);              // tile_dim1 (tile_dim2=0)
  g1[5] = (int)stride0;                        // tensor_dim0_stride lo32
  g1[6] = 0;                                   // stride hi | dim1_stride lo
  g1[7] = 0;
  i32x4 z4 = {0, 0, 0, 0};
#if __clang_major__ >= 23
  i32x8 z8 = {0, 0, 0, 0, 0, 0, 0, 0};
  __builtin_amdgcn_tensor_load_to_lds(g0, g1, z4, z4, z8, 0);
#else
  __builtin_amdgcn_tensor_load_to_lds(g0, g1, z4, z4, 0);
#endif
}
__device__ inline u32 lds_addr_of(const void* p) {
  return (u32)(unsigned long long)p;
}
#endif

// ---------------------------------------------------------------------------
// Tiled GEMM: Out[M,N] = A[M,K] @ B[K,N] + bias[N]
//   A_BF16:   A is bf16 (else fp32, converted on LDS fill)
//   OUT_MODE: 0 = fp32 row-major [M][N]
//             1 = bf16 head-split  [h][M][64]   (h = n/64)     (Q, K)
//             2 = bf16 head-split transposed [h*64+d][M] i.e. [n][M]  (V^T)
// Block: 128 threads (4 waves), tile 128x64, K-step 32.
// ---------------------------------------------------------------------------
template <bool A_BF16, int OUT_MODE>
__global__ __launch_bounds__(128) void gemm_bias_kernel(
    const void* __restrict__ Ap, const float* __restrict__ Bp,
    const float* __restrict__ bias, void* __restrict__ Outp,
    int M, int N, int K) {
  __shared__ alignas(16) bf16 As[128 * 32];  // [row][k]
  __shared__ alignas(16) bf16 Bs[64 * 32];   // [n][k] (transposed)

  int tid  = threadIdx.x;
  int wave = tid >> 5, lane = tid & 31;
  int half = lane >> 4, ln = lane & 15;
  int m0 = blockIdx.x * 128;
  int n0 = blockIdx.y * 64;

  f32x8 acc[2][4] = {};

  for (int k0 = 0; k0 < K; k0 += 32) {
    __syncthreads();
    if (A_BF16) {
      // 128 rows x 32 bf16 = 512 uint4
#pragma unroll 2
      for (int t = tid; t < 128 * 4; t += 128) {
        int r = t >> 2, c8 = t & 3;
        ((uint4*)As)[t] =
            ((const uint4*)((const bf16*)Ap + (size_t)(m0 + r) * K + k0))[c8];
      }
    } else {
      // 128 rows x 32 fp32 = 1024 float4, convert to bf16, 64-bit LDS stores
#pragma unroll 4
      for (int t = tid; t < 128 * 8; t += 128) {
        int r = t >> 3, c4 = t & 7;
        float4 v = ((const float4*)((const float*)Ap + (size_t)(m0 + r) * K + k0))[c4];
        union { uint2 u; bf16 h[4]; } pk;
        pk.h[0] = (bf16)v.x; pk.h[1] = (bf16)v.y;
        pk.h[2] = (bf16)v.z; pk.h[3] = (bf16)v.w;
        ((uint2*)(As + r * 32))[c4] = pk.u;
      }
    }
    // B: 32 k x 64 n fp32 = 512 float4; store transposed [n][k]
#pragma unroll 4
    for (int t = tid; t < 32 * 16; t += 128) {
      int kk = t >> 4, n4 = (t & 15) * 4;
      float4 v = ((const float4*)(Bp + (size_t)(k0 + kk) * N + n0))[t & 15];
      Bs[(n4 + 0) * 32 + kk] = (bf16)v.x;
      Bs[(n4 + 1) * 32 + kk] = (bf16)v.y;
      Bs[(n4 + 2) * 32 + kk] = (bf16)v.z;
      Bs[(n4 + 3) * 32 + kk] = (bf16)v.w;
    }
    if (k0 + 32 < K) {  // prefetch next k-tile into L2 (global_prefetch_b8)
      if (A_BF16)
        __builtin_prefetch((const bf16*)Ap + (size_t)(m0 + tid) * K + k0 + 32, 0, 3);
      else
        __builtin_prefetch((const float*)Ap + (size_t)(m0 + tid) * K + k0 + 32, 0, 3);
      __builtin_prefetch(Bp + (size_t)(k0 + 32 + (tid & 31)) * N + n0 + ((tid >> 5) << 4), 0, 3);
    }
    __syncthreads();

    bf16x16 af[2];
#pragma unroll
    for (int mi = 0; mi < 2; ++mi)
      af[mi] = load_a_frag(As, wave * 32 + mi * 16, 0, 32);
#pragma unroll
    for (int ni = 0; ni < 4; ++ni) {
      bf16x16 bfr = load_b_frag(Bs, ni * 16, 0, 32);
#pragma unroll
      for (int mi = 0; mi < 2; ++mi)
        acc[mi][ni] = wmma_bf16(af[mi], bfr, acc[mi][ni]);
    }
  }

#pragma unroll
  for (int mi = 0; mi < 2; ++mi)
#pragma unroll
    for (int ni = 0; ni < 4; ++ni)
#pragma unroll
      for (int i = 0; i < 8; ++i) {
        int row = m0 + wave * 32 + mi * 16 + i + half * 8;
        int col = n0 + ni * 16 + ln;
        float v = acc[mi][ni][i] + bias[col];
        if (OUT_MODE == 1) {
          int h = col >> 6, d = col & 63;
          ((bf16*)Outp)[((size_t)h * M + row) * DK + d] = (bf16)v;
        } else if (OUT_MODE == 2) {
          ((bf16*)Outp)[(size_t)col * M + row] = (bf16)v;  // [h][dk][S]
        } else {
          ((float*)Outp)[(size_t)row * N + col] = v;
        }
      }
}

// ---------------------------------------------------------------------------
// Flash attention with T5 relative-position bias + causal mask.
// Grid: (SEQ/64, NHEADS). Block: 128 threads = 4 waves, each wave owns 16
// query rows. K tiles [key][dk] and V tiles [dk][key] are double-buffered in
// LDS; when available, the Tensor Data Mover prefetches the next tile pair
// (TENSORcnt-tracked) while all waves run WMMA on the current one.
// ---------------------------------------------------------------------------
__global__ __launch_bounds__(128) void flash_attn_kernel(
    const bf16* __restrict__ Qb, const bf16* __restrict__ Kb,
    const bf16* __restrict__ Vb, const float* __restrict__ rpe,
    bf16* __restrict__ AO) {
  __shared__ alignas(16) bf16 Qs[64 * 64];        // [qrow][dk]
  __shared__ alignas(16) bf16 Ksb[2][64 * 64];    // [key][dk]
  __shared__ alignas(16) bf16 Vtb[2][64 * 64];    // [dk][key]
  __shared__ alignas(16) bf16 Ps[4 * 16 * 64];    // per-wave P tile [row][key]
  __shared__ float rpe_s[32];

  int h = blockIdx.y, qt = blockIdx.x;
  int tid  = threadIdx.x;
  int wave = tid >> 5, lane = tid & 31;
  int half = lane >> 4, ln = lane & 15;

  const bf16* Qh = Qb + (size_t)h * SEQ * DK;
  const bf16* Kh = Kb + (size_t)h * SEQ * DK;   // [S][dk]
  const bf16* Vh = Vb + (size_t)h * DK * SEQ;   // [dk][S] (pre-transposed)

  if (tid < 32) rpe_s[tid] = rpe[tid * NHEADS + h];

  int q0 = qt * 64;
  // Q tile: 64 rows x 64 bf16 = 512 uint4
  for (int t = tid; t < 64 * 8; t += 128)
    ((uint4*)Qs)[t] = ((const uint4*)(Qh + (size_t)(q0 + (t >> 3)) * DK))[t & 7];

#if HAVE_TDM
  if (tid < 32) {  // wave 0 issues DMA for tile kt=0 into buffer 0
    tdm_load_tile_2d(lds_addr_of(&Ksb[0][0]), Kh, DK, SEQ, 64, 64, DK);
    tdm_load_tile_2d(lds_addr_of(&Vtb[0][0]), Vh, SEQ, DK, 64, 64, SEQ);
  }
#endif
  __syncthreads();

  bf16x16 qa[2];
  qa[0] = load_a_frag(Qs, wave * 16, 0, 64);
  qa[1] = load_a_frag(Qs, wave * 16, 32, 64);

  f32x8 oacc[4] = {};
  float mrow[8], lrow[8];
#pragma unroll
  for (int i = 0; i < 8; ++i) { mrow[i] = -1e30f; lrow[i] = 0.0f; }

  int qrow = q0 + wave * 16;
  bf16* Pw = Ps + wave * 16 * 64;

  for (int kt = 0; kt <= qt; ++kt) {
    int k0 = kt * 64;
    int cur = kt & 1;
    const bf16* Ks = Ksb[cur];
    const bf16* Vt = Vtb[cur];

#if HAVE_TDM
    if (tid < 32) {
      if (kt + 1 <= qt) {  // prefetch next tile pair into the other buffer
        tdm_load_tile_2d(lds_addr_of(&Ksb[cur ^ 1][0]),
                         Kh + (size_t)(k0 + 64) * DK, DK, SEQ, 64, 64, DK);
        tdm_load_tile_2d(lds_addr_of(&Vtb[cur ^ 1][0]),
                         Vh + (size_t)(k0 + 64), SEQ, DK, 64, 64, SEQ);
        __builtin_amdgcn_s_wait_tensorcnt(2);  // current pair done (in-order)
      } else {
        __builtin_amdgcn_s_wait_tensorcnt(0);
      }
    }
    __syncthreads();  // current tile visible to all waves
#else
    __syncthreads();  // previous iteration's readers done
    for (int t = tid; t < 64 * 8; t += 128) {
      ((uint4*)Ksb[cur])[t] =
          ((const uint4*)(Kh + (size_t)(k0 + (t >> 3)) * DK))[t & 7];
      ((uint4*)Vtb[cur])[t] =
          ((const uint4*)(Vh + (size_t)(t >> 3) * SEQ + k0))[t & 7];
    }
    __syncthreads();
#endif

    // S = Q @ K^T  (16x64 per wave)
    f32x8 sacc[4];
#pragma unroll
    for (int n = 0; n < 4; ++n) {
      f32x8 c = {};
#pragma unroll
      for (int kk = 0; kk < 2; ++kk) {
        bf16x16 bfr = load_b_frag(Ks, n * 16, kk * 32, 64);
        c = wmma_bf16(qa[kk], bfr, c);
      }
      sacc[n] = c;
    }

    // scale + T5 bias + causal mask, tracking per-row tile max
    float p[4][8], tmax[8];
#pragma unroll
    for (int i = 0; i < 8; ++i) tmax[i] = -1e30f;
#pragma unroll
    for (int n = 0; n < 4; ++n)
#pragma unroll
      for (int i = 0; i < 8; ++i) {
        int q = qrow + i + half * 8;
        int k = k0 + n * 16 + ln;
        float s = sacc[n][i] * 0.125f;  // 1/sqrt(64)
        int nd = q - k; if (nd < 0) nd = 0;
        int bucket;
        if (nd < 16) bucket = nd;
        else {
          // 16 + floor(log(n/16)/log(256) * 16) == 16 + floor(2*log2(n/16))
          int b = 16 + (int)(2.0f * __log2f((float)nd * 0.0625f));
          bucket = b > 31 ? 31 : b;
        }
        s += rpe_s[bucket];
        if (k > q) s = -1e30f;  // causal
        p[n][i] = s;
        tmax[i] = fmaxf(tmax[i], s);
      }
    // row max across the 16 lanes of each half-wave
#pragma unroll
    for (int off = 1; off < 16; off <<= 1)
#pragma unroll
      for (int i = 0; i < 8; ++i) tmax[i] = fmaxf(tmax[i], __shfl_xor(tmax[i], off));

    float alpha[8], tsum[8];
#pragma unroll
    for (int i = 0; i < 8; ++i) {
      float mn = fmaxf(mrow[i], tmax[i]);
      alpha[i] = __expf(mrow[i] - mn);
      mrow[i] = mn;
      tsum[i] = 0.0f;
    }
#pragma unroll
    for (int n = 0; n < 4; ++n)
#pragma unroll
      for (int i = 0; i < 8; ++i) {
        float pe = __expf(p[n][i] - mrow[i]);
        p[n][i] = pe;
        tsum[i] += pe;
      }
#pragma unroll
    for (int off = 1; off < 16; off <<= 1)
#pragma unroll
      for (int i = 0; i < 8; ++i) tsum[i] += __shfl_xor(tsum[i], off);
#pragma unroll
    for (int i = 0; i < 8; ++i) lrow[i] = lrow[i] * alpha[i] + tsum[i];
#pragma unroll
    for (int n = 0; n < 4; ++n)
#pragma unroll
      for (int i = 0; i < 8; ++i) oacc[n][i] *= alpha[i];

    // P (D-layout) -> LDS -> A-layout for P@V
#pragma unroll
    for (int n = 0; n < 4; ++n)
#pragma unroll
      for (int i = 0; i < 8; ++i)
        Pw[(i + half * 8) * 64 + n * 16 + ln] = (bf16)p[n][i];
    __syncthreads();

    bf16x16 pa[2];
    pa[0] = load_a_frag(Pw, 0, 0, 64);
    pa[1] = load_a_frag(Pw, 0, 32, 64);
#pragma unroll
    for (int n = 0; n < 4; ++n)
#pragma unroll
      for (int kk = 0; kk < 2; ++kk) {
        bf16x16 bfr = load_b_frag(Vt, n * 16, kk * 32, 64);
        oacc[n] = wmma_bf16(pa[kk], bfr, oacc[n]);
      }
#if HAVE_TDM
    __syncthreads();  // all reads of `cur` done before it is refilled
#endif
  }

  // normalize and write attention output in [S][DMODEL] bf16 layout
#pragma unroll
  for (int n = 0; n < 4; ++n)
#pragma unroll
    for (int i = 0; i < 8; ++i) {
      int q = qrow + i + half * 8;
      int d = n * 16 + ln;
      float v = oacc[n][i] / lrow[i];
      AO[(size_t)q * DMODEL + h * DK + d] = (bf16)v;
    }
}

// ---------------------------------------------------------------------------
extern "C" void kernel_launch(void* const* d_in, const int* in_sizes, int n_in,
                              void* d_out, int out_size, void* d_ws, size_t ws_size,
                              hipStream_t stream) {
  (void)in_sizes; (void)n_in; (void)out_size; (void)ws_size;
  const float* query = (const float*)d_in[0];
  const float* key_  = (const float*)d_in[1];
  const float* value = (const float*)d_in[2];
  // d_in[3] = causal mask: implemented analytically in-kernel
  const float* Wq = (const float*)d_in[4];
  const float* bq = (const float*)d_in[5];
  const float* Wk = (const float*)d_in[6];
  const float* bk = (const float*)d_in[7];
  const float* Wv = (const float*)d_in[8];
  const float* bv = (const float*)d_in[9];
  const float* Wo = (const float*)d_in[10];
  const float* bo = (const float*)d_in[11];
  const float* rpe = (const float*)d_in[12];

  char* ws = (char*)d_ws;
  bf16* Qb = (bf16*)(ws);                        // [H][S][64] bf16, 8 MB
  bf16* Kb = (bf16*)(ws + ((size_t)8  << 20));   // [H][S][64] bf16, 8 MB
  bf16* Vb = (bf16*)(ws + ((size_t)16 << 20));   // [H][64][S] bf16 (V^T), 8 MB
  bf16* AO = (bf16*)(ws + ((size_t)24 << 20));   // [S][D]    bf16, 8 MB

  dim3 gg(SEQ / 128, DMODEL / 64);
  gemm_bias_kernel<false, 1><<<gg, 128, 0, stream>>>(query, Wq, bq, Qb, SEQ, DMODEL, DMODEL);
  gemm_bias_kernel<false, 1><<<gg, 128, 0, stream>>>(key_,  Wk, bk, Kb, SEQ, DMODEL, DMODEL);
  gemm_bias_kernel<false, 2><<<gg, 128, 0, stream>>>(value, Wv, bv, Vb, SEQ, DMODEL, DMODEL);

  flash_attn_kernel<<<dim3(SEQ / 64, NHEADS), 128, 0, stream>>>(Qb, Kb, Vb, rpe, AO);

  gemm_bias_kernel<true, 0><<<gg, 128, 0, stream>>>(AO, Wo, bo, d_out, SEQ, DMODEL, DMODEL);
}
